// TinyGPT_9277129359924
// MI455X (gfx1250) — matrix operand order repaired and checked by reading, not compiled, not run on hardware
//
#include <hip/hip_runtime.h>

// ---------------------------------------------------------------------------
// TinyGPT forward on gfx1250 (MI455X). Mixed precision: weights pre-transposed
// to [N,K] f16, all GEMMs via v_wmma_f32_16x16x32_f16 (f32 accumulate),
// double-buffered LDS staging with async global->LDS copies when available.
// ---------------------------------------------------------------------------

typedef _Float16 hlf;
typedef __attribute__((ext_vector_type(16))) _Float16 v16h;
typedef __attribute__((ext_vector_type(8)))  _Float16 h8;
typedef __attribute__((ext_vector_type(8)))  float    v8f;

enum { OUT_F16 = 0, OUT_RELU = 1, OUT_RESID = 2, OUT_F32 = 3 };

// -------- async global->LDS (CDNA5), guarded; fallback = sync copy ----------
// __has_builtin acts as the "this compiler supports gfx1250 async ops" probe;
// the copy itself is inline asm (ISA 15.18.3 operand order: lds-addr VGPR,
// 64-bit global address, no saddr) to avoid builtin-signature differences.
#if defined(__has_builtin)
#if __has_builtin(__builtin_amdgcn_global_load_async_to_lds_b128)
#define TGPT_ASYNC 1
#endif
#endif
#ifndef TGPT_ASYNC
#define TGPT_ASYNC 0
#endif

__device__ __forceinline__ void async_cp16(hlf* lds, const hlf* glob) {
#if TGPT_ASYNC
  // Low 32 bits of a generic LDS pointer == LDS byte offset (ISA aperture rule).
  unsigned int       loff  = (unsigned int)(unsigned long long)lds;
  unsigned long long gaddr = (unsigned long long)glob;
  asm volatile("global_load_async_to_lds_b128 %0, %1, off"
               :
               : "v"(loff), "v"(gaddr)
               : "memory");
#else
  *(h8*)lds = *(const h8*)glob;
#endif
}

template <int N>
__device__ __forceinline__ void wait_async_le() {
#if TGPT_ASYNC
#if __has_builtin(__builtin_amdgcn_s_wait_asynccnt)
  __builtin_amdgcn_s_wait_asynccnt(N);
#else
  asm volatile("s_wait_asynccnt %0" ::"i"(N) : "memory");
#endif
#endif
}

// Build a 16-half WMMA fragment from two contiguous 16-byte LDS chunks.
__device__ __forceinline__ v16h ld_frag(const hlf* p0, const hlf* p1) {
  union { v16h v; h8 h[2]; } u;
  u.h[0] = *(const h8*)p0;
  u.h[1] = *(const h8*)p1;
  return u.v;
}

// ---------------------------------------------------------------------------
// Tiled WMMA GEMM:  C[z] = epilogue(A[z] @ Bt[z]^T + bias)
//   A : [M,K] f16 row-major (row stride lda)
//   Bt: [N,K] f16 row-major (row stride ldb)  -- i.e. B stored transposed, so
//       staging is pure contiguous 16B copies (no LDS scatter anywhere).
//   8 waves (wave32), wave grid 4x2; BK=32; double-buffered LDS + async copy.
// ---------------------------------------------------------------------------
template <int BM, int BN, int MODE>
__global__ __launch_bounds__(256) void gemm_wmma(
    const hlf* __restrict__ A, int lda, long sAb, long sAh,
    const hlf* __restrict__ Bt, int ldb, long sBb, long sBh,
    const float* __restrict__ bias,
    void* __restrict__ C, int ldc, long sCb, long sCh,
    int K, int nheads, int causal)
{
  constexpr int BK = 32;
  constexpr int WM = BM / 4, WN = BN / 2;
  constexpr int MT = WM / 16, NT = WN / 16;
  constexpr int AIT = (BM * 4) / 256;     // 16B chunks per thread for A tile
  constexpr int BIT = (BN * 4) / 256;     // 16B chunks per thread for B tile
  constexpr int BATCH = AIT + BIT;        // per-wave async ops per tile

  const int t    = threadIdx.x;
  const int lane = t & 31;
  const int w    = t >> 5;
  const int wm   = w & 3, wn = w >> 2;
  const int lrow = lane & 15;
  const int hsel = lane >> 4;
  const int m0 = blockIdx.y * BM;
  const int n0 = blockIdx.x * BN;
  if (causal && n0 >= m0 + BM) return;    // block fully above causal diagonal

  const int z  = blockIdx.z;
  const int bi = z / nheads, hi = z % nheads;
  const hlf* Ab = A  + (long)bi * sAb + (long)hi * sAh;
  const hlf* Bb = Bt + (long)bi * sBb + (long)hi * sBh;
  const long coff = (long)bi * sCb + (long)hi * sCh;

  // +8 halves pad: 80B row stride (16B aligned, bank-conflict friendly)
  __shared__ alignas(16) hlf As[2][BM][BK + 8];
  __shared__ alignas(16) hlf Bs[2][BN][BK + 8];

  v8f acc[MT][NT];
#pragma unroll
  for (int i = 0; i < MT; ++i)
#pragma unroll
    for (int j = 0; j < NT; ++j)
#pragma unroll
      for (int e = 0; e < 8; ++e) acc[i][j][e] = 0.f;

  auto stage = [&](int kt, int buf) {
    const int k0 = kt * BK;
#pragma unroll
    for (int i = 0; i < AIT; ++i) {
      int c = t + i * 256;
      int r = c >> 2, kc = (c & 3) * 8;
      async_cp16(&As[buf][r][kc], &Ab[(long)(m0 + r) * lda + k0 + kc]);
    }
#pragma unroll
    for (int i = 0; i < BIT; ++i) {
      int c = t + i * 256;
      int r = c >> 2, kc = (c & 3) * 8;
      async_cp16(&Bs[buf][r][kc], &Bb[(long)(n0 + r) * ldb + k0 + kc]);
    }
  };

  const int nk = K / BK;
  stage(0, 0);
  for (int kt = 0; kt < nk; ++kt) {
    const int buf = kt & 1;
    if (kt + 1 < nk) {
      stage(kt + 1, buf ^ 1);   // prefetch next tile into other buffer
      wait_async_le<BATCH>();   // my copies for *this* tile have landed
    } else {
      wait_async_le<0>();
    }
    __syncthreads();            // everyone's copies for this tile have landed

    v16h af[MT], bf[NT];
#pragma unroll
    for (int i = 0; i < MT; ++i) {
      const hlf* p = &As[buf][wm * WM + i * 16 + lrow][hsel * 8];
      af[i] = ld_frag(p, p + 16);
    }
#pragma unroll
    for (int j = 0; j < NT; ++j) {
      const hlf* p = &Bs[buf][wn * WN + j * 16 + lrow][hsel * 8];
      bf[j] = ld_frag(p, p + 16);
    }
#pragma unroll
    for (int i = 0; i < MT; ++i)
#pragma unroll
      for (int j = 0; j < NT; ++j)
        acc[i][j] = __builtin_amdgcn_wmma_f32_16x16x32_f16(
            false, af[i], false, bf[j], (short)0, acc[i][j], false, false);

    __syncthreads();            // done reading buf before it is overwritten
  }

  // ---- epilogue: C/D layout lane=(col 0..15), VGPR r -> row r + 8*hsel ----
#pragma unroll
  for (int i = 0; i < MT; ++i) {
#pragma unroll
    for (int j = 0; j < NT; ++j) {
      const int gn = n0 + wn * WN + j * 16 + lrow;
      const float bvv = bias ? bias[gn] : 0.f;
#pragma unroll
      for (int r = 0; r < 8; ++r) {
        const int gm = m0 + wm * WM + i * 16 + hsel * 8 + r;
        const long cix = coff + (long)gm * ldc + gn;
        float v = acc[i][j][r] + bvv;
        if constexpr (MODE == OUT_F16) {
          ((hlf*)C)[cix] = (hlf)v;
        } else if constexpr (MODE == OUT_RELU) {
          ((hlf*)C)[cix] = (hlf)fmaxf(v, 0.f);
        } else if constexpr (MODE == OUT_RESID) {
          ((float*)C)[cix] += v;
        } else {
          ((float*)C)[cix] = v;
        }
      }
    }
  }
}

// ---------------------------------------------------------------------------
// Transpose-convert: in f32 [rows,cols] -> out f16 [cols,rows]. z = matrix id.
// ---------------------------------------------------------------------------
__global__ __launch_bounds__(256) void trans_cvt_f32_f16(
    const float* __restrict__ in, hlf* __restrict__ out, int rows, int cols)
{
  __shared__ float tile[32][33];
  const long base = (long)blockIdx.z * rows * cols;
  const int r0 = blockIdx.y * 32, c0 = blockIdx.x * 32;
  const int tx = threadIdx.x & 31, ty = threadIdx.x >> 5;
#pragma unroll
  for (int i = 0; i < 4; ++i)
    tile[ty + i * 8][tx] = in[base + (long)(r0 + ty + i * 8) * cols + c0 + tx];
  __syncthreads();
#pragma unroll
  for (int i = 0; i < 4; ++i)
    out[base + (long)(c0 + ty + i * 8) * rows + r0 + tx] = (hlf)tile[tx][ty + i * 8];
}

// ---------------------------------------------------------------------------
// Per-head transpose of V: [B*S, D] head-sliced -> vt[bh][HD][S]  (f16)
// ---------------------------------------------------------------------------
__global__ __launch_bounds__(256) void trans_v_f16(
    const hlf* __restrict__ vsrc, hlf* __restrict__ vt)
{
  __shared__ hlf tile[32][33];
  const int bh = blockIdx.z, bi = bh / 12, hi = bh % 12;
  const int s0 = blockIdx.x * 32, d0 = blockIdx.y * 32;
  const int tx = threadIdx.x & 31, ty = threadIdx.x >> 5;
  const hlf* src = vsrc + (long)bi * 1024 * 768 + hi * 64;
#pragma unroll
  for (int i = 0; i < 4; ++i)
    tile[ty + i * 8][tx] = src[(long)(s0 + ty + i * 8) * 768 + d0 + tx];
  __syncthreads();
  hlf* dst = vt + (long)bh * 64 * 1024;
#pragma unroll
  for (int i = 0; i < 4; ++i)
    dst[(long)(d0 + ty + i * 8) * 1024 + s0 + tx] = tile[tx][ty + i * 8];
}

// ---------------------------------------------------------------------------
// x[m,:] = tok_emb[idx[m],:] + pos_emb[m % S,:]   (fp32)
// ---------------------------------------------------------------------------
__global__ __launch_bounds__(256) void embed_kernel(
    const int* __restrict__ idx, const float* __restrict__ tok,
    const float* __restrict__ pos, float* __restrict__ x)
{
  long g = (long)blockIdx.x * 256 + threadIdx.x;
  int m = (int)(g / 768);
  int d = (int)(g % 768);
  int s = m & 1023;                      // S = 1024
  x[g] = tok[(long)idx[m] * 768 + d] + pos[(long)s * 768 + d];
}

// ---------------------------------------------------------------------------
// LayerNorm over D=768, fp32 in -> f16 out. One 256-thread block per row.
// ---------------------------------------------------------------------------
__global__ __launch_bounds__(256) void layernorm_to_f16(
    const float* __restrict__ x, const float* __restrict__ g,
    const float* __restrict__ b, hlf* __restrict__ out)
{
  const int D = 768;
  const int row = blockIdx.x, t = threadIdx.x;
  const float* xr = x + (long)row * D;
  float v0 = xr[t], v1 = xr[t + 256], v2 = xr[t + 512];
  __shared__ float r1[256], r2[256];
  r1[t] = v0 + v1 + v2;
  r2[t] = v0 * v0 + v1 * v1 + v2 * v2;
  __syncthreads();
  for (int st = 128; st > 0; st >>= 1) {
    if (t < st) { r1[t] += r1[t + st]; r2[t] += r2[t + st]; }
    __syncthreads();
  }
  const float mu = r1[0] * (1.f / D);
  const float var = r2[0] * (1.f / D) - mu * mu;
  const float rs = rsqrtf(var + 1e-5f);
  hlf* o = out + (long)row * D;
  o[t]       = (hlf)((v0 - mu) * rs * g[t]       + b[t]);
  o[t + 256] = (hlf)((v1 - mu) * rs * g[t + 256] + b[t + 256]);
  o[t + 512] = (hlf)((v2 - mu) * rs * g[t + 512] + b[t + 512]);
}

// ---------------------------------------------------------------------------
// Causal softmax over one score row (length S=1024); writes P as f16,
// zero-fills masked tail so P@V needs no masking. scale = 1/sqrt(HD).
// ---------------------------------------------------------------------------
__global__ __launch_bounds__(256) void softmax_causal(
    const float* __restrict__ sc, hlf* __restrict__ P, float scale)
{
  const int S = 1024;
  const int q = blockIdx.x, t = threadIdx.x;
  const long base = ((long)blockIdx.y * S + q) * (long)S;
  const int valid = q + 1;
  __shared__ float red[256];

  float m = -3.0e38f;
  for (int k = t; k < valid; k += 256) m = fmaxf(m, sc[base + k]);
  red[t] = m; __syncthreads();
  for (int st = 128; st > 0; st >>= 1) {
    if (t < st) red[t] = fmaxf(red[t], red[t + st]);
    __syncthreads();
  }
  m = red[0];
  __syncthreads();

  float sum = 0.f;
  for (int k = t; k < valid; k += 256) {
    float e = __expf(scale * (sc[base + k] - m));
    P[base + k] = (hlf)e;                // unnormalized, in (0,1]
    sum += e;
  }
  red[t] = sum; __syncthreads();
  for (int st = 128; st > 0; st >>= 1) {
    if (t < st) red[t] += red[t + st];
    __syncthreads();
  }
  const float inv = 1.f / red[0];

  for (int k = t; k < S; k += 256) {
    if (k < valid) P[base + k] = (hlf)((float)P[base + k] * inv);
    else           P[base + k] = (hlf)0.f;
  }
}

// ---------------------------------------------------------------------------
extern "C" void kernel_launch(void* const* d_in, const int* in_sizes, int n_in,
                              void* d_out, int out_size, void* d_ws, size_t ws_size,
                              hipStream_t stream) {
  (void)in_sizes; (void)n_in; (void)out_size; (void)ws_size;
  constexpr int Bn = 2, S = 1024, D = 768, H = 12, HD = 64, L = 6,
                DFF = 3072, V = 32000;
  constexpr int M = Bn * S;              // 2048

  const int*   idx  = (const int*)  d_in[0];
  const float* tok  = (const float*)d_in[1];
  const float* pos  = (const float*)d_in[2];
  const float* ln1w = (const float*)d_in[3];
  const float* ln1b = (const float*)d_in[4];
  const float* Wq   = (const float*)d_in[5];
  const float* bq   = (const float*)d_in[6];
  const float* Wk   = (const float*)d_in[7];
  const float* bk   = (const float*)d_in[8];
  const float* Wv   = (const float*)d_in[9];
  const float* bv   = (const float*)d_in[10];
  const float* Wo   = (const float*)d_in[11];
  const float* bo   = (const float*)d_in[12];
  const float* ln2w = (const float*)d_in[13];
  const float* ln2b = (const float*)d_in[14];
  const float* W1   = (const float*)d_in[15];
  const float* b1   = (const float*)d_in[16];
  const float* W2   = (const float*)d_in[17];
  const float* b2   = (const float*)d_in[18];
  const float* lnfw = (const float*)d_in[19];
  const float* lnfb = (const float*)d_in[20];
  const float* Wlm  = (const float*)d_in[21];

  // ---- workspace carve-out ----
  char* ws = (char*)d_ws;
  size_t off = 0;
  auto alloc = [&](size_t bytes) -> void* {
    void* p = ws + off;
    off += (bytes + 255) & ~(size_t)255;
    return p;
  };
  hlf* WqT  = (hlf*)alloc((size_t)L * D * D * sizeof(hlf));    // [N,K] per layer
  hlf* WkT  = (hlf*)alloc((size_t)L * D * D * sizeof(hlf));
  hlf* WvT  = (hlf*)alloc((size_t)L * D * D * sizeof(hlf));
  hlf* WoT  = (hlf*)alloc((size_t)L * D * D * sizeof(hlf));
  hlf* W1T  = (hlf*)alloc((size_t)L * D * DFF * sizeof(hlf));  // [DFF,D]
  hlf* W2T  = (hlf*)alloc((size_t)L * DFF * D * sizeof(hlf));  // [D,DFF]
  hlf* WlmT = (hlf*)alloc((size_t)D * V * sizeof(hlf));        // [V,D]
  float* x      = (float*)alloc((size_t)M * D * sizeof(float));
  hlf*   h16    = (hlf*)  alloc((size_t)M * D * sizeof(hlf));
  hlf*   q16    = (hlf*)  alloc((size_t)M * D * sizeof(hlf));
  hlf*   k16    = (hlf*)  alloc((size_t)M * D * sizeof(hlf));
  hlf*   v16    = (hlf*)  alloc((size_t)M * D * sizeof(hlf));
  hlf*   vt16   = (hlf*)  alloc((size_t)Bn * H * HD * S * sizeof(hlf));
  hlf*   y16    = (hlf*)  alloc((size_t)M * D * sizeof(hlf));
  hlf*   ffn16  = (hlf*)  alloc((size_t)M * DFF * sizeof(hlf));
  float* scores = (float*)alloc((size_t)Bn * H * S * S * sizeof(float));
  hlf*   P16    = (hlf*)  alloc((size_t)Bn * H * S * S * sizeof(hlf));

  const dim3 blk(256);

  // ---- one-time (per call) weight transpose+convert to f16 [N,K] ----
  trans_cvt_f32_f16<<<dim3(D / 32, D / 32, L), blk, 0, stream>>>(Wq, WqT, D, D);
  trans_cvt_f32_f16<<<dim3(D / 32, D / 32, L), blk, 0, stream>>>(Wk, WkT, D, D);
  trans_cvt_f32_f16<<<dim3(D / 32, D / 32, L), blk, 0, stream>>>(Wv, WvT, D, D);
  trans_cvt_f32_f16<<<dim3(D / 32, D / 32, L), blk, 0, stream>>>(Wo, WoT, D, D);
  trans_cvt_f32_f16<<<dim3(DFF / 32, D / 32, L), blk, 0, stream>>>(W1, W1T, D, DFF);
  trans_cvt_f32_f16<<<dim3(D / 32, DFF / 32, L), blk, 0, stream>>>(W2, W2T, DFF, D);
  trans_cvt_f32_f16<<<dim3(V / 32, D / 32, 1), blk, 0, stream>>>(Wlm, WlmT, D, V);

  // ---- embeddings ----
  embed_kernel<<<dim3(M * D / 256), blk, 0, stream>>>(idx, tok, pos, x);

  const long sQb = (long)S * D;          // batch stride for q/k/y slices
  const long sPb = (long)H * S * S;      // batch stride for scores/P
  const long sVb = (long)H * HD * S;     // batch stride for vt

  for (int l = 0; l < L; ++l) {
    // attn: h = LN1(x)
    layernorm_to_f16<<<dim3(M), blk, 0, stream>>>(x, ln1w + (long)l * D, ln1b + (long)l * D, h16);
    // q,k,v = h@W + b  (f16 out)
    gemm_wmma<128, 128, OUT_F16><<<dim3(D / 128, M / 128, 1), blk, 0, stream>>>(
        h16, D, 0, 0, WqT + (long)l * D * D, D, 0, 0, bq + (long)l * D,
        (void*)q16, D, 0, 0, D, 1, 0);
    gemm_wmma<128, 128, OUT_F16><<<dim3(D / 128, M / 128, 1), blk, 0, stream>>>(
        h16, D, 0, 0, WkT + (long)l * D * D, D, 0, 0, bk + (long)l * D,
        (void*)k16, D, 0, 0, D, 1, 0);
    gemm_wmma<128, 128, OUT_F16><<<dim3(D / 128, M / 128, 1), blk, 0, stream>>>(
        h16, D, 0, 0, WvT + (long)l * D * D, D, 0, 0, bv + (long)l * D,
        (void*)v16, D, 0, 0, D, 1, 0);
    // scores[b,h] = Q_bh @ K_bh^T  (f32; K already has the [N,K] layout)
    gemm_wmma<128, 128, OUT_F32><<<dim3(S / 128, S / 128, Bn * H), blk, 0, stream>>>(
        q16, D, sQb, HD, k16, D, sQb, HD, nullptr,
        (void*)scores, S, sPb, (long)S * S, HD, H, 1);
    // P = causal softmax(scale * scores)  (f16)
    softmax_causal<<<dim3(S, Bn * H), blk, 0, stream>>>(scores, P16, 0.125f);
    // vt[bh] = V_bh^T  (so P@V also uses the fast [N,K] staging)
    trans_v_f16<<<dim3(S / 32, HD / 32, Bn * H), blk, 0, stream>>>(v16, vt16);
    // y[b,h] = P @ V_bh  (f16, written into head slice of y)
    gemm_wmma<128, 64, OUT_F16><<<dim3(HD / 64, S / 128, Bn * H), blk, 0, stream>>>(
        P16, S, sPb, (long)S * S, vt16, S, sVb, (long)HD * S, nullptr,
        (void*)y16, D, sQb, HD, S, H, 0);
    // x += y@Wo + bo
    gemm_wmma<128, 128, OUT_RESID><<<dim3(D / 128, M / 128, 1), blk, 0, stream>>>(
        y16, D, 0, 0, WoT + (long)l * D * D, D, 0, 0, bo + (long)l * D,
        (void*)x, D, 0, 0, D, 1, 0);
    // MLP: h = LN2(x); ffn = relu(h@W1 + b1); x += ffn@W2 + b2
    layernorm_to_f16<<<dim3(M), blk, 0, stream>>>(x, ln2w + (long)l * D, ln2b + (long)l * D, h16);
    gemm_wmma<128, 128, OUT_RELU><<<dim3(DFF / 128, M / 128, 1), blk, 0, stream>>>(
        h16, D, 0, 0, W1T + (long)l * D * DFF, D, 0, 0, b1 + (long)l * DFF,
        (void*)ffn16, DFF, 0, 0, D, 1, 0);
    gemm_wmma<128, 128, OUT_RESID><<<dim3(D / 128, M / 128, 1), blk, 0, stream>>>(
        ffn16, DFF, 0, 0, W2T + (long)l * DFF * D, DFF, 0, 0, b2 + (long)l * D,
        (void*)x, D, 0, 0, DFF, 1, 0);
  }

  // logits = LNf(x) @ lm_head  (f32 out)
  layernorm_to_f16<<<dim3(M), blk, 0, stream>>>(x, lnfw, lnfb, h16);
  gemm_wmma<128, 128, OUT_F32><<<dim3(V / 128, M / 128, 1), blk, 0, stream>>>(
      h16, D, 0, 0, WlmT, D, 0, 0, nullptr,
      (void*)d_out, V, 0, 0, D, 1, 0);
}
